// NSHNet_63247688401698
// MI455X (gfx1250) — matrix operand order repaired and checked by reading, not compiled, not run on hardware
//
#include <hip/hip_runtime.h>
#include <math.h>

// ---- problem constants (match reference) ----
constexpr int LL     = 2;        // layers
constexpr int NEXP   = 4;        // experts
constexpr int HDIM   = 768;      // hidden
constexpr int FFD    = 4 * HDIM; // 3072
constexpr int NHEADS = HDIM / 64;// 12
constexpr int DHEAD  = 64;
constexpr int BB     = 8;
constexpr int SS     = 256;
constexpr int TT     = BB * SS;  // 2048 tokens
constexpr float LN_EPS = 1e-5f;

typedef __attribute__((ext_vector_type(16))) _Float16 v16h;
typedef __attribute__((ext_vector_type(8)))  _Float16 v8h;
typedef __attribute__((ext_vector_type(8)))  float    v8f;

// ======================= WMMA core =======================
__device__ __forceinline__ v8f wmma_16x16x32(v16h a, v16h b, v8f c) {
  return __builtin_amdgcn_wmma_f32_16x16x32_f16(
      /*neg_a=*/false, a, /*neg_b=*/false, b,
      /*c_mod=*/(short)0, c, /*reuse_a=*/false, /*reuse_b=*/false);
}

__device__ __forceinline__ v16h join8(v8h lo, v8h hi) {
  return __builtin_shufflevector(lo, hi, 0, 1, 2, 3, 4, 5, 6, 7,
                                 8, 9, 10, 11, 12, 13, 14, 15);
}

// ---- CDNA5 async global->LDS copy (16 bytes/lane, ASYNCcnt-tracked) ----
__device__ __forceinline__ void async_copy_b128(uint32_t lds_off, const void* gsrc) {
  const unsigned long long ga = (unsigned long long)gsrc;
  asm volatile("global_load_async_to_lds_b128 %0, %1, off"
               :: "v"(lds_off), "v"(ga) : "memory");
}
__device__ __forceinline__ void wait_async_le(int n) {
  if (n == 0) asm volatile("s_wait_asynccnt 0" ::: "memory");
  else        asm volatile("s_wait_asynccnt 3" ::: "memory");
}

// ---- f16 direct-from-global fragment loaders (attention GEMMs) ----
__device__ __forceinline__ v16h frag_a_rows(const _Float16* __restrict__ base, int ld) {
  const int lane = threadIdx.x & 31;
  const _Float16* ap = base + (size_t)(lane & 15) * ld;
  const int kb8 = (lane >> 4) << 3;
  return join8(*(const v8h*)(ap + kb8), *(const v8h*)(ap + 16 + kb8));
}
__device__ __forceinline__ v16h frag_b_rows(const _Float16* __restrict__ base, int ld) {
  const int lane = threadIdx.x & 31;
  const _Float16* bp = base + (size_t)(lane & 15) * ld + ((lane >> 4) << 3);
  return join8(*(const v8h*)(bp), *(const v8h*)(bp + 16));
}

// ======================= LDS-blocked, double-buffered GEMM (f16 operands) =======================
// 256 threads = 8 waves; macro-tile 128(M) x 64(N); wave tile 32x32; K-step 32.
// Async global->LDS staging is pipelined: prefetch tile i+1 while computing tile i.
// A : f16 [M][K] row-major (AMODE: 0 plain, 1 perm-gathered guarded, 2 guarded at seg0)
// Bt: f16 [N][K] row-major (pre-transposed weights)
// ACT: 0 none, 1 exact GELU ; CM: 0 plain store, 1 scatter via perm + residual from Xres
// OT : 0 store f32 to C32, 1 store f16 to C16
template <int AMODE, int ACT, int CM, int OT>
__global__ void k_gemm_tiled(const _Float16* __restrict__ A, int lda,
                             const _Float16* __restrict__ Bt, int ldbt,
                             const float* __restrict__ bias,
                             float* __restrict__ C32, _Float16* __restrict__ C16,
                             int ldc, int K, int Nn,
                             const int* __restrict__ perm,
                             const int* __restrict__ off,
                             const float* __restrict__ Xres,
                             const _Float16* __restrict__ zsrc) {
  __shared__ _Float16 Alds[2][128 * 32];   // [buf][m][k], 2 x 8 KB
  __shared__ _Float16 Blds[2][64 * 32];    // [buf][n][k], 2 x 4 KB

  const int e = blockIdx.z;
  int seg0 = 0, mcnt = 1 << 30;
  if (AMODE != 0) {
    seg0 = off[e];
    mcnt = off[e + 1] - seg0;
  }
  const int m_base = blockIdx.y * 128;
  if (AMODE != 0 && m_base >= mcnt) return;
  const int gn0 = blockIdx.x * 64;

  const _Float16* Bte  = (AMODE == 0) ? Bt   : Bt   + (size_t)e * Nn * K;
  const float*   biase = (AMODE == 0) ? bias : bias + (size_t)e * Nn;

  const int t = threadIdx.x;
  const int lane = t & 31;
  const int w = t >> 5;
  const int mw = (w >> 1) * 32;
  const int nw = (w & 1) * 32;
  const int fm = lane & 15;
  const int kb8 = (lane >> 4) << 3;
  const int fn = lane & 15;
  const int kb16 = (lane >> 4) << 4;

  // Per-thread staging descriptors (row/part constant across K-steps).
  const int arow0 = t >> 2, apart0 = t & 3;             // chunk t
  const int arow1 = (t + 256) >> 2, apart1 = t & 3;     // chunk t+256
  const int brow = t >> 2, bpart = t & 3;

  // Precompute per-thread A source row pointers (gather/guard resolved once).
  const _Float16* asrc0;
  const _Float16* asrc1;
  {
    const int r0 = m_base + arow0, r1 = m_base + arow1;
    if (AMODE == 0) {
      asrc0 = A + (size_t)r0 * lda;
      asrc1 = A + (size_t)r1 * lda;
    } else if (AMODE == 1) {
      asrc0 = (r0 < mcnt) ? (A + (size_t)perm[seg0 + r0] * lda) : nullptr;
      asrc1 = (r1 < mcnt) ? (A + (size_t)perm[seg0 + r1] * lda) : nullptr;
    } else {
      asrc0 = (r0 < mcnt) ? (A + (size_t)(seg0 + r0) * lda) : nullptr;
      asrc1 = (r1 < mcnt) ? (A + (size_t)(seg0 + r1) * lda) : nullptr;
    }
  }
  const _Float16* bsrc = Bte + (size_t)(gn0 + brow) * ldbt;

  const uint32_t aB0 = (uint32_t)(uintptr_t)(&Alds[0][0]);
  const uint32_t aB1 = (uint32_t)(uintptr_t)(&Alds[1][0]);
  const uint32_t bB0 = (uint32_t)(uintptr_t)(&Blds[0][0]);
  const uint32_t bB1 = (uint32_t)(uintptr_t)(&Blds[1][0]);

  // Issue 3 async b128 copies for tile at k0 into buffer `buf`.
  auto stage = [&](int k0, int buf) {
    const uint32_t aBase = buf ? aB1 : aB0;
    const uint32_t bBase = buf ? bB1 : bB0;
    async_copy_b128(aBase + (uint32_t)t * 16,
                    asrc0 ? (asrc0 + k0 + apart0 * 8) : zsrc);
    async_copy_b128(aBase + (uint32_t)(t + 256) * 16,
                    asrc1 ? (asrc1 + k0 + apart1 * 8) : zsrc);
    async_copy_b128(bBase + (uint32_t)t * 16, bsrc + k0 + bpart * 8);
  };

  v8f acc[2][2] = {};

  const int nsteps = K / 32;
  stage(0, 0);
  for (int it = 0; it < nsteps; ++it) {
    const int cur = it & 1;
    if (it + 1 < nsteps) {
      stage((it + 1) * 32, cur ^ 1);   // prefetch next tile into other buffer
      wait_async_le(3);                // tile `it` complete (in-order retirement)
    } else {
      wait_async_le(0);
    }
    __syncthreads();

    // ---- fragments from LDS (ds_load_b128) ----
    v16h af[2], bf[2];
#pragma unroll
    for (int s = 0; s < 2; ++s) {
      const _Float16* ap = &Alds[cur][(mw + s * 16 + fm) * 32];
      af[s] = join8(*(const v8h*)(ap + kb8), *(const v8h*)(ap + 16 + kb8));
      const _Float16* bp = &Blds[cur][(nw + s * 16 + fn) * 32 + kb8];
      bf[s] = join8(*(const v8h*)(bp), *(const v8h*)(bp + 16));
    }
#pragma unroll
    for (int sm = 0; sm < 2; ++sm)
#pragma unroll
      for (int sn = 0; sn < 2; ++sn)
        acc[sm][sn] = wmma_16x16x32(af[sm], bf[sn], acc[sm][sn]);
    __syncthreads();   // protects buffer reused at it+2
  }

  // ---- store ----
  const int mb = (lane >> 4) << 3;
#pragma unroll
  for (int sm = 0; sm < 2; ++sm) {
#pragma unroll
    for (int sn = 0; sn < 2; ++sn) {
      const int gn = gn0 + nw + sn * 16 + fn;
      const float bv = biase[gn];
#pragma unroll
      for (int i = 0; i < 8; ++i) {
        const int row_local = m_base + mw + sm * 16 + mb + i;
        if (AMODE != 0 && row_local >= mcnt) continue;
        float val = acc[sm][sn][i] + bv;
        if (ACT == 1) val = 0.5f * val * (1.0f + erff(val * 0.70710678118654752f));
        if (CM == 0) {
          const size_t crow = (AMODE == 1) ? (size_t)(seg0 + row_local) : (size_t)row_local;
          if (OT == 1) C16[crow * ldc + gn] = (_Float16)val;
          else         C32[crow * ldc + gn] = val;
        } else {
          const int tok = perm[seg0 + row_local];
          C32[(size_t)tok * ldc + gn] = Xres[(size_t)tok * ldc + gn] + val;
        }
      }
    }
  }
}

// ======================= conversion / pack kernels =======================

// src f32 [K][N] -> dst f16 [N][K]  (weight transpose-convert, once per launch)
__global__ void k_cvt_t(const float* __restrict__ src, _Float16* __restrict__ dst,
                        int K, int N) {
  const size_t idx = (size_t)blockIdx.x * 256 + threadIdx.x;
  if (idx >= (size_t)K * N) return;
  const int k = (int)(idx / N);
  const int n = (int)(idx % N);
  dst[(size_t)n * K + k] = (_Float16)src[idx];
}

// Vh f16 [t][HDIM] -> Vt f16 [(b*NH+h)*DHEAD + d][SS]
__global__ void k_pack_vt(const _Float16* __restrict__ Vh, _Float16* __restrict__ Vt) {
  const int idx = blockIdx.x * blockDim.x + threadIdx.x;
  if (idx >= BB * NHEADS * DHEAD * SS) return;
  const int s = idx & (SS - 1);
  const int d = (idx >> 8) & (DHEAD - 1);
  const int z = idx >> 14;              // b*NHEADS + h
  const int b = z / NHEADS, h = z % NHEADS;
  Vt[idx] = Vh[(size_t)(b * SS + s) * HDIM + h * DHEAD + d];
}

// ======================= elementwise / reduction kernels =======================

__global__ void k_add_pe(const float* __restrict__ xin, float* __restrict__ xout) {
  const int idx = blockIdx.x * blockDim.x + threadIdx.x;
  if (idx >= TT * HDIM) return;
  const int d = idx % HDIM;
  const int s = (idx / HDIM) % SS;
  const int pair = d >> 1;
  const float div = expf((float)(2 * pair) * (-9.210340371976184f / (float)HDIM));
  const float ang = (float)s * div;
  const float pe = (d & 1) ? cosf(ang) : sinf(ang);
  xout[idx] = xin[idx] + pe;
}

// LayerNorm, f32 in -> f16 out (feeds WMMA GEMMs)
__global__ void k_layernorm(const float* __restrict__ x, const float* __restrict__ w,
                            const float* __restrict__ b, _Float16* __restrict__ y) {
  const int tk = blockIdx.x;
  const float* row = x + (size_t)tk * HDIM;
  _Float16* out = y + (size_t)tk * HDIM;
  __shared__ float red[256];
  const int c = threadIdx.x;
  const float v0 = row[c], v1 = row[c + 256], v2 = row[c + 512];
  red[c] = v0 + v1 + v2;
  __syncthreads();
  for (int off = 128; off > 0; off >>= 1) {
    if (c < off) red[c] += red[c + off];
    __syncthreads();
  }
  const float mu = red[0] / (float)HDIM;
  __syncthreads();
  const float d0 = v0 - mu, d1 = v1 - mu, d2 = v2 - mu;
  red[c] = d0 * d0 + d1 * d1 + d2 * d2;
  __syncthreads();
  for (int off = 128; off > 0; off >>= 1) {
    if (c < off) red[c] += red[c + off];
    __syncthreads();
  }
  const float rs = rsqrtf(red[0] / (float)HDIM + LN_EPS);
  out[c]       = (_Float16)(d0 * rs * w[c]       + b[c]);
  out[c + 256] = (_Float16)(d1 * rs * w[c + 256] + b[c + 256]);
  out[c + 512] = (_Float16)(d2 * rs * w[c + 512] + b[c + 512]);
}

// softmax over f32 scores row, masked; writes f16 probs (A operand of ctx GEMM)
__global__ void k_softmax_mask(const float* __restrict__ scores,
                               const int* __restrict__ note_pos,
                               _Float16* __restrict__ probs) {
  const int row = blockIdx.x;         // (b*NHEADS + h)*SS + q
  const int z = row / SS;
  const int bb = z / NHEADS;
  const float* p = scores + (size_t)row * SS;
  _Float16* ph = probs + (size_t)row * SS;
  const int col = threadIdx.x;
  const float msk = (note_pos[bb * SS + col] != 0) ? 0.0f : -10000.0f;
  const float vloc = p[col] + msk;
  __shared__ float red[256];
  red[col] = vloc;
  __syncthreads();
  for (int off = 128; off > 0; off >>= 1) {
    if (col < off) red[col] = fmaxf(red[col], red[col + off]);
    __syncthreads();
  }
  const float mx = red[0];
  __syncthreads();
  const float e = expf(vloc - mx);
  red[col] = e;
  __syncthreads();
  for (int off = 128; off > 0; off >>= 1) {
    if (col < off) red[col] += red[col + off];
    __syncthreads();
  }
  ph[col] = (_Float16)(e / red[0]);
}

// Deterministic stable expert routing.
__global__ void k_route(const int* __restrict__ ntype, int* __restrict__ perm,
                        int* __restrict__ off) {
  __shared__ int cnt[NEXP + 1];
  const int e = threadIdx.x;
  if (e < NEXP) {
    int c = 0;
    for (int t = 0; t < TT; ++t) c += (ntype[t] == e) ? 1 : 0;
    cnt[e + 1] = c;
  }
  __syncthreads();
  if (e == 0) {
    cnt[0] = 0;
    for (int i = 0; i < NEXP; ++i) cnt[i + 1] += cnt[i];
    for (int i = 0; i <= NEXP; ++i) off[i] = cnt[i];
  }
  __syncthreads();
  if (e < NEXP) {
    int w = cnt[e];
    for (int t = 0; t < TT; ++t)
      if (ntype[t] == e) perm[w++] = t;
  }
}

__global__ void k_scatter(const float* __restrict__ X2, const int* __restrict__ note_pos,
                          float* __restrict__ Xout) {
  const int idx = blockIdx.x * blockDim.x + threadIdx.x;
  if (idx >= TT * HDIM) return;
  const int t = idx / HDIM, d = idx % HDIM;
  const int src = note_pos[t];
  Xout[idx] = (src == 0) ? 0.0f : X2[(size_t)(src - 1) * HDIM + d];
}

// ======================= attention WMMA kernels (f16 operands) =======================

// scores[z][q][k] = Q_bh @ K_bh^T  (Qh/Kh f16 row-major [t][HDIM])
__global__ void k_scores(const _Float16* __restrict__ qh, const _Float16* __restrict__ kh,
                         float* __restrict__ scores) {
  const int z = blockIdx.z;
  const int bb = z / NHEADS, hh = z % NHEADS;
  const int n0 = blockIdx.x * 16;  // key index
  const int m0 = blockIdx.y * 16;  // query index
  const _Float16* qb = qh + (size_t)bb * SS * HDIM + hh * DHEAD;
  const _Float16* kb = kh + (size_t)bb * SS * HDIM + hh * DHEAD;
  v8f c = {};
#pragma unroll
  for (int k0 = 0; k0 < DHEAD; k0 += 32) {
    v16h a = frag_a_rows(qb + (size_t)m0 * HDIM + k0, HDIM);
    v16h b = frag_b_rows(kb + (size_t)n0 * HDIM + k0, HDIM);
    c = wmma_16x16x32(a, b, c);
  }
  float* out = scores + (size_t)z * SS * SS;
  const int lane = threadIdx.x & 31;
  const int n = lane & 15;
  const int mb = (lane >> 4) << 3;
#pragma unroll
  for (int i = 0; i < 8; ++i)
    out[(size_t)(m0 + mb + i) * SS + n0 + n] = c[i];
}

// X += P @ V  (P f16 [z][q][k], Vt f16 [z*DHEAD + d][s]; residual via accumulator init)
__global__ void k_ctx_residual(const _Float16* __restrict__ probs,
                               const _Float16* __restrict__ vt,
                               float* __restrict__ X) {
  const int z = blockIdx.z;
  const int bb = z / NHEADS, hh = z % NHEADS;
  const int n0 = blockIdx.x * 16;  // within head dim
  const int m0 = blockIdx.y * 16;  // query index
  const _Float16* P  = probs + (size_t)z * SS * SS;
  const _Float16* vb = vt + ((size_t)z * DHEAD) * SS;   // rows: d, cols: s
  float* xb = X + (size_t)bb * SS * HDIM + hh * DHEAD;
  const int lane = threadIdx.x & 31;
  const int n = lane & 15;
  const int mb = (lane >> 4) << 3;
  v8f c;
#pragma unroll
  for (int i = 0; i < 8; ++i)
    c[i] = xb[(size_t)(m0 + mb + i) * HDIM + n0 + n];
  for (int k0 = 0; k0 < SS; k0 += 32) {
    v16h a = frag_a_rows(P + (size_t)m0 * SS + k0, SS);
    v16h b = frag_b_rows(vb + (size_t)n0 * SS + k0, SS);
    c = wmma_16x16x32(a, b, c);
  }
#pragma unroll
  for (int i = 0; i < 8; ++i)
    xb[(size_t)(m0 + mb + i) * HDIM + n0 + n] = c[i];
}

// ======================= host orchestration =======================

extern "C" void kernel_launch(void* const* d_in, const int* in_sizes, int n_in,
                              void* d_out, int out_size, void* d_ws, size_t ws_size,
                              hipStream_t stream) {
  (void)in_sizes; (void)n_in; (void)out_size; (void)ws_size;
  const float* x      = (const float*)d_in[0];
  const int* note_pos = (const int*)d_in[1];
  const int* ntype    = (const int*)d_in[2];
  const float* Wq = (const float*)d_in[3];  const float* bq = (const float*)d_in[4];
  const float* Wk = (const float*)d_in[5];  const float* bk = (const float*)d_in[6];
  const float* Wv = (const float*)d_in[7];  const float* bv = (const float*)d_in[8];
  const float* ln1w = (const float*)d_in[9];  const float* ln1b = (const float*)d_in[10];
  const float* ln2w = (const float*)d_in[11]; const float* ln2b = (const float*)d_in[12];
  const float* W1 = (const float*)d_in[13]; const float* b1 = (const float*)d_in[14];
  const float* W2 = (const float*)d_in[15]; const float* b2 = (const float*)d_in[16];

  // ---- workspace layout (f32 first, then f16, then ints) ----
  float* wsf = (float*)d_ws;
  float* X   = wsf; wsf += (size_t)TT * HDIM;
  float* X2  = wsf; wsf += (size_t)TT * HDIM;
  float* Sc  = wsf; wsf += (size_t)BB * NHEADS * SS * SS;
  float* zp  = wsf; wsf += 64;                    // zero page for guarded async loads
  _Float16* wsh = (_Float16*)wsf;
  _Float16* Hln = wsh; wsh += (size_t)TT * HDIM;
  _Float16* Qh  = wsh; wsh += (size_t)TT * HDIM;
  _Float16* Kh  = wsh; wsh += (size_t)TT * HDIM;
  _Float16* Vh  = wsh; wsh += (size_t)TT * HDIM;
  _Float16* Vt  = wsh; wsh += (size_t)BB * NHEADS * DHEAD * SS;
  _Float16* Ph  = wsh; wsh += (size_t)BB * NHEADS * SS * SS;
  _Float16* Hb  = wsh; wsh += (size_t)TT * FFD;
  _Float16* Wqt = wsh; wsh += (size_t)HDIM * HDIM;
  _Float16* Wkt = wsh; wsh += (size_t)HDIM * HDIM;
  _Float16* Wvt = wsh; wsh += (size_t)HDIM * HDIM;
  _Float16* W1t = wsh; wsh += (size_t)NEXP * HDIM * FFD;
  _Float16* W2t = wsh; wsh += (size_t)NEXP * FFD * HDIM;
  int* perm = (int*)wsh;
  int* off  = perm + TT;

  const dim3 wave(32);
  const dim3 blk256(256);

  hipMemsetAsync(zp, 0, 64 * sizeof(float), stream);
  k_add_pe<<<(TT * HDIM + 255) / 256, 256, 0, stream>>>(x, X);
  k_route<<<1, 32, 0, stream>>>(ntype, perm, off);

  for (int l = 0; l < LL; ++l) {
    // ---- one-time f16 transpose-convert of this layer's weights ----
    const int nHH = (HDIM * HDIM + 255) / 256;
    k_cvt_t<<<nHH, 256, 0, stream>>>(Wq + (size_t)l * HDIM * HDIM, Wqt, HDIM, HDIM);
    k_cvt_t<<<nHH, 256, 0, stream>>>(Wk + (size_t)l * HDIM * HDIM, Wkt, HDIM, HDIM);
    k_cvt_t<<<nHH, 256, 0, stream>>>(Wv + (size_t)l * HDIM * HDIM, Wvt, HDIM, HDIM);
    const int nHF = (HDIM * FFD + 255) / 256;
    for (int e = 0; e < NEXP; ++e) {
      k_cvt_t<<<nHF, 256, 0, stream>>>(W1 + ((size_t)l * NEXP + e) * HDIM * FFD,
                                       W1t + (size_t)e * FFD * HDIM, HDIM, FFD);
      k_cvt_t<<<nHF, 256, 0, stream>>>(W2 + ((size_t)l * NEXP + e) * FFD * HDIM,
                                       W2t + (size_t)e * HDIM * FFD, FFD, HDIM);
    }

    k_layernorm<<<TT, 256, 0, stream>>>(X, ln1w + (size_t)l * HDIM, ln1b + (size_t)l * HDIM, Hln);

    // QKV projections (M=2048, N=768, K=768), f16 out
    const dim3 gq(HDIM / 64, TT / 128, 1);
    k_gemm_tiled<0, 0, 0, 1><<<gq, blk256, 0, stream>>>(
        Hln, HDIM, Wqt, HDIM, bq + (size_t)l * HDIM, nullptr, Qh,
        HDIM, HDIM, HDIM, perm, off, nullptr, (const _Float16*)zp);
    k_gemm_tiled<0, 0, 0, 1><<<gq, blk256, 0, stream>>>(
        Hln, HDIM, Wkt, HDIM, bk + (size_t)l * HDIM, nullptr, Kh,
        HDIM, HDIM, HDIM, perm, off, nullptr, (const _Float16*)zp);
    k_gemm_tiled<0, 0, 0, 1><<<gq, blk256, 0, stream>>>(
        Hln, HDIM, Wvt, HDIM, bv + (size_t)l * HDIM, nullptr, Vh,
        HDIM, HDIM, HDIM, perm, off, nullptr, (const _Float16*)zp);

    k_pack_vt<<<(BB * NHEADS * DHEAD * SS + 255) / 256, 256, 0, stream>>>(Vh, Vt);

    const dim3 gs(SS / 16, SS / 16, BB * NHEADS);
    k_scores<<<gs, wave, 0, stream>>>(Qh, Kh, Sc);
    k_softmax_mask<<<BB * NHEADS * SS, 256, 0, stream>>>(Sc, note_pos, Ph);
    const dim3 gc(DHEAD / 16, SS / 16, BB * NHEADS);
    k_ctx_residual<<<gc, wave, 0, stream>>>(Ph, Vt, X);

    k_layernorm<<<TT, 256, 0, stream>>>(X, ln2w + (size_t)l * HDIM, ln2b + (size_t)l * HDIM, Hln);

    // Expert FFN (compacted): FFN1 f16 out + GELU, FFN2 f32 + residual + scatter-store
    const dim3 g1(FFD / 64, TT / 128, NEXP);
    k_gemm_tiled<1, 1, 0, 1><<<g1, blk256, 0, stream>>>(
        Hln, HDIM, W1t, HDIM, b1 + (size_t)l * NEXP * FFD, nullptr, Hb,
        FFD, HDIM, FFD, perm, off, nullptr, (const _Float16*)zp);
    const dim3 g2(HDIM / 64, TT / 128, NEXP);
    k_gemm_tiled<2, 0, 1, 0><<<g2, blk256, 0, stream>>>(
        Hb, FFD, W2t, FFD, b2 + (size_t)l * NEXP * HDIM, X2, nullptr,
        HDIM, FFD, HDIM, perm, off, X, (const _Float16*)zp);

    k_scatter<<<(TT * HDIM + 255) / 256, 256, 0, stream>>>(X2, note_pos, X);
  }

  hipMemcpyAsync(d_out, X, (size_t)TT * HDIM * sizeof(float),
                 hipMemcpyDeviceToDevice, stream);
}
